// Generator_67929202753658
// MI455X (gfx1250) — matrix operand order repaired and checked by reading, not compiled, not run on hardware
//
#include <hip/hip_runtime.h>

typedef __attribute__((ext_vector_type(16))) _Float16 v16h;
typedef __attribute__((ext_vector_type(8)))  _Float16 v8h;
typedef __attribute__((ext_vector_type(8)))  float    v8f;
typedef __attribute__((ext_vector_type(4)))  int      v4i;

#define KGEMM     4608          // 9 * 512
#define NGEMM     16384         // 16 * 32 * 32
#define NKB       144           // KGEMM / 32
#define INV_SQRT_STYLE 0.04419417382415922f   // 1/sqrt(512)
#define WSCALE    0.014731391274719742f        // 1/sqrt(512*9)
#define SQRT2F    1.4142135623730951f
#define KPAD      40            // LDS row stride in f16 elems (80B, 16B-aligned, conflict-free)

// ---- gfx1250 async global->LDS path (guarded; falls back to sync copy) ----
#if defined(__gfx1250__) && __has_builtin(__builtin_amdgcn_global_load_async_to_lds_b128)
#define HAVE_ASYNC_LDS 1
#endif

// pointer-to-vector-in-addrspace typedefs (attribute applies to pointee)
typedef __attribute__((address_space(1))) v4i* gptr_v4i;
typedef __attribute__((address_space(3))) v4i* lptr_v4i;

__device__ __forceinline__ void stage16(const _Float16* g, _Float16* l) {
#if HAVE_ASYNC_LDS
  __builtin_amdgcn_global_load_async_to_lds_b128((gptr_v4i)g, (lptr_v4i)l, 0, 0);
#else
  *(v8h*)l = *(const v8h*)g;
#endif
}

__device__ __forceinline__ void wait_async_le4() {
#if HAVE_ASYNC_LDS
#if __has_builtin(__builtin_amdgcn_s_wait_asynccnt)
  __builtin_amdgcn_s_wait_asynccnt(4);
#else
  asm volatile("s_wait_asynccnt 0x4" ::: "memory");
#endif
#endif
}

__device__ __forceinline__ void wait_async_le0() {
#if HAVE_ASYNC_LDS
#if __has_builtin(__builtin_amdgcn_s_wait_asynccnt)
  __builtin_amdgcn_s_wait_asynccnt(0);
#else
  asm volatile("s_wait_asynccnt 0x0" ::: "memory");
#endif
#endif
}

// ---------------- prep kernels ----------------

// s[b,i] = style[b,:] . mod_w[i,:] / sqrt(512) + mod_b[i]
__global__ void k_modulate(const float* __restrict__ style,
                           const float* __restrict__ mod_w,
                           const float* __restrict__ mod_b,
                           float* __restrict__ s) {
  int gid = blockIdx.x * 256 + threadIdx.x;     // 0..8191
  int b = gid >> 9, i = gid & 511;
  const float* st = style + b * 512;
  const float* mw = mod_w + i * 512;
  float acc = 0.f;
  for (int k = 0; k < 512; ++k) acc += st[k] * mw[k];
  s[gid] = acc * INV_SQRT_STYLE + mod_b[i];
}

// wsum[o,i] = sum_{ky,kx} cw[o,i,ky,kx]^2
__global__ void k_wsum(const float* __restrict__ cw, float* __restrict__ wsum) {
  int gid = blockIdx.x * 256 + threadIdx.x;     // 0..262143  (o*512+i)
  const float* p = cw + (size_t)gid * 9;
  float a = 0.f;
  for (int t = 0; t < 9; ++t) { float v = p[t]; a += v * v; }
  wsum[gid] = a;
}

// dscale[b,o] = wscale * rsqrt(wscale^2 * sum_i s[b,i]^2 * wsum[o,i] + eps)
__global__ void k_dscale(const float* __restrict__ s, const float* __restrict__ wsum,
                         float* __restrict__ dscale) {
  int gid = blockIdx.x * 256 + threadIdx.x;     // 0..8191  (b*512+o)
  int b = gid >> 9, o = gid & 511;
  const float* sb = s + (b << 9);
  const float* wo = wsum + (o << 9);
  float a = 0.f;
  for (int i = 0; i < 512; ++i) { float sv = sb[i]; a += sv * sv * wo[i]; }
  dscale[gid] = WSCALE * rsqrtf(WSCALE * WSCALE * a + 1e-8f);
}

// Composite (blur o convT) phase weights.
// W6[m] = sum_{u+k=m} blur[u]*wflip[k] ; phase (py,px) picks 3x3 taps of W6.
// Layout: weff[ph][o][k], k = (dy*3+dx)*512 + i  (K-major, contiguous per tap)
__global__ void k_weff(const float* __restrict__ cw, _Float16* __restrict__ weff) {
  int gid = blockIdx.x * 256 + threadIdx.x;     // 0..262143  (o*512+i)
  int o = gid >> 9, i = gid & 511;
  const float* p = cw + (size_t)gid * 9;
  float wf[3][3];
  for (int ky = 0; ky < 3; ++ky)
    for (int kx = 0; kx < 3; ++kx)
      wf[ky][kx] = p[(2 - ky) * 3 + (2 - kx)];   // spatially flipped kernel
  const float kb1[4] = {1.f, 3.f, 3.f, 1.f};
  float W6[6][6];
  for (int my = 0; my < 6; ++my)
    for (int mx = 0; mx < 6; ++mx) {
      float a = 0.f;
      for (int ky = 0; ky < 3; ++ky) {
        int uy = my - ky; if (uy < 0 || uy > 3) continue;
        for (int kx = 0; kx < 3; ++kx) {
          int ux = mx - kx; if (ux < 0 || ux > 3) continue;
          a += wf[ky][kx] * kb1[uy] * kb1[ux] * (1.f / 16.f);
        }
      }
      W6[my][mx] = a;
    }
  for (int ph = 0; ph < 4; ++ph) {
    int py = ph >> 1, px = ph & 1;
    for (int dy = 0; dy < 3; ++dy)
      for (int dx = 0; dx < 3; ++dx) {
        int my = 2 * dy + 1 - py;
        int mx = 2 * dx + 1 - px;
        weff[((size_t)(ph * 512 + o)) * KGEMM + (dy * 3 + dx) * 512 + i] =
            (_Float16)W6[my][mx];
      }
  }
}

// xs[b][hp][wp][i] (NHWC, 34x34 zero-padded) = f16( x[b,i,hp-1,wp-1] * s[b,i] )
__global__ void k_xs(const float* __restrict__ x, const float* __restrict__ s,
                     _Float16* __restrict__ xs) {
  size_t gid = (size_t)blockIdx.x * 256 + threadIdx.x;  // 0..9469951
  int i = (int)(gid & 511);
  size_t r = gid >> 9;
  int wp = (int)(r % 34); r /= 34;
  int hp = (int)(r % 34);
  int b  = (int)(r / 34);
  _Float16 v = (_Float16)0.f;
  if (hp >= 1 && hp <= 32 && wp >= 1 && wp <= 32) {
    float xv = x[(((size_t)(b * 512 + i)) * 32 + (hp - 1)) * 32 + (wp - 1)];
    v = (_Float16)(xv * s[(b << 9) + i]);
  }
  xs[gid] = v;
}

// ---------------- main WMMA implicit-GEMM conv ----------------
// grid (N/128, M/128, 4 phases), 256 threads = 8 waves in 2(M)x4(N).
// Double-buffered LDS, async global->LDS staging (ASYNCcnt), 8 WMMAs/iter/wave.
__global__ __launch_bounds__(256)
void k_conv_wmma(const _Float16* __restrict__ weff, const _Float16* __restrict__ xs,
                 const float* __restrict__ dscale, const float* __restrict__ noise,
                 const float* __restrict__ noise_w, const float* __restrict__ act_bias,
                 float* __restrict__ out) {
  __shared__ __align__(16) _Float16 As[2][128 * KPAD];
  __shared__ __align__(16) _Float16 Bs[2][128 * KPAD];

  const int tid  = threadIdx.x;
  const int n0   = blockIdx.x * 128;
  const int m0   = blockIdx.y * 128;
  const int ph   = blockIdx.z;
  const int py   = ph >> 1, px = ph & 1;
  const int lane = tid & 31, wid = tid >> 5;
  const int wm   = wid >> 2, wn = wid & 3;
  const int half = lane >> 4, nl = lane & 15;

  // staging geometry: this thread stages rows {arow, arow+64}, 16B chunk 'apart'
  const int arow = tid >> 2, apart = tid & 3;
  const size_t weffbase = (size_t)(ph * 512 + m0) * KGEMM;
  const _Float16* pA = weff + weffbase + (size_t)arow * KGEMM + apart * 8;
  const int ldoff = arow * KPAD + apart * 8;

  const int nr0 = n0 + arow, nr1 = n0 + arow + 64;
  const int b0 = nr0 >> 10, a0 = (nr0 >> 5) & 31, c0 = nr0 & 31;
  const int b1 = nr1 >> 10, a1 = (nr1 >> 5) & 31, c1 = nr1 & 31;

  v8f acc[4][2] = {};

  auto stage = [&](int kb, int buf) {
    const int t  = kb >> 4;            // tap 0..8
    const int kk = kb & 15;            // 32-ch sub-block within tap
    const int dy = t / 3, dx = t - 3 * dy;
    const _Float16* gA = pA + (t << 9) + (kk << 5);
    stage16(gA,                        &As[buf][ldoff]);
    stage16(gA + (size_t)64 * KGEMM,   &As[buf][ldoff + 64 * KPAD]);
    const _Float16* gB0 = xs + ((size_t)((b0 * 34 + a0 + dy) * 34 + (c0 + dx)) << 9)
                             + (kk << 5) + apart * 8;
    const _Float16* gB1 = xs + ((size_t)((b1 * 34 + a1 + dy) * 34 + (c1 + dx)) << 9)
                             + (kk << 5) + apart * 8;
    stage16(gB0, &Bs[buf][ldoff]);
    stage16(gB1, &Bs[buf][ldoff + 64 * KPAD]);
  };

  stage(0, 0);
  for (int kb = 0; kb < NKB; ++kb) {
    const int buf = kb & 1;
    if (kb + 1 < NKB) {
      stage(kb + 1, buf ^ 1);   // prefetch next tile into other buffer
      wait_async_le4();         // 4 oldest outstanding (this tile) have landed
    } else {
      wait_async_le0();
    }
    __syncthreads();

    // B fragments: lane group holds contiguous K half (k = 16*half + s, same order as A)
    v16h bf[2];
#pragma unroll
    for (int nt = 0; nt < 2; ++nt) {
      const int nrow = wn * 32 + nt * 16 + nl;
      v8h lo = *(const v8h*)&Bs[buf][nrow * KPAD + half * 16];
      v8h hi = *(const v8h*)&Bs[buf][nrow * KPAD + half * 16 + 8];
      bf[nt] = __builtin_shufflevector(lo, hi, 0,1,2,3,4,5,6,7,8,9,10,11,12,13,14,15);
    }
    // A fragments: K = {g*16..g*16+15} (contiguous K half, matches B)
#pragma unroll
    for (int mt = 0; mt < 4; ++mt) {
      const int mrow = wm * 64 + mt * 16 + nl;
      v8h lo = *(const v8h*)&As[buf][mrow * KPAD + half * 16];
      v8h hi = *(const v8h*)&As[buf][mrow * KPAD + half * 16 + 8];
      v16h af = __builtin_shufflevector(lo, hi, 0,1,2,3,4,5,6,7,8,9,10,11,12,13,14,15);
#pragma unroll
      for (int nt = 0; nt < 2; ++nt)
        acc[mt][nt] = __builtin_amdgcn_wmma_f32_16x16x32_f16(
            false, af, false, bf[nt], (short)0, acc[mt][nt], false, false);
    }
    __syncthreads();
  }

  // epilogue: scale by dscale(b,o), add noise + bias, leaky(0.2)*sqrt(2)
  const float nwv = noise_w[0];
#pragma unroll
  for (int nt = 0; nt < 2; ++nt) {
    const int n = n0 + wn * 32 + nt * 16 + nl;
    const int b = n >> 10, a = (n >> 5) & 31, c = n & 31;
    const int y = 2 * a + py, xx = 2 * c + px;
    const float nz = nwv * noise[(b << 12) + (y << 6) + xx];
    const size_t obase = ((size_t)b) << 9;
#pragma unroll
    for (int mt = 0; mt < 4; ++mt) {
#pragma unroll
      for (int v = 0; v < 8; ++v) {
        const int o = m0 + wm * 64 + mt * 16 + 8 * half + v;
        float val = acc[mt][nt][v] * dscale[obase + o] + nz + act_bias[o];
        val = (val > 0.f ? val : 0.2f * val) * SQRT2F;
        out[((obase + o) << 12) + (y << 6) + xx] = val;
      }
    }
  }
}

// ---------------- launch ----------------

extern "C" void kernel_launch(void* const* d_in, const int* in_sizes, int n_in,
                              void* d_out, int out_size, void* d_ws, size_t ws_size,
                              hipStream_t stream) {
  (void)in_sizes; (void)n_in; (void)out_size; (void)ws_size;
  const float* x       = (const float*)d_in[0];
  const float* style   = (const float*)d_in[1];
  const float* noise   = (const float*)d_in[2];
  const float* cw      = (const float*)d_in[3];
  const float* mod_w   = (const float*)d_in[4];
  const float* mod_b   = (const float*)d_in[5];
  const float* noise_w = (const float*)d_in[6];
  const float* act_b   = (const float*)d_in[7];
  float* out = (float*)d_out;

  char* ws = (char*)d_ws;
  float*     s      = (float*)(ws + 0);              //  32 KB (16x512)
  float*     dscale = (float*)(ws + 32768);          //  32 KB (16x512)
  float*     wsum   = (float*)(ws + 65536);          //   1 MB (512x512)
  _Float16*  weff   = (_Float16*)(ws + 1114112);     // 18.9 MB (4x512x4608)
  _Float16*  xs     = (_Float16*)(ws + 19988480);    // 18.9 MB (16x34x34x512)

  k_modulate<<<dim3(32),    dim3(256), 0, stream>>>(style, mod_w, mod_b, s);
  k_wsum    <<<dim3(1024),  dim3(256), 0, stream>>>(cw, wsum);
  k_weff    <<<dim3(1024),  dim3(256), 0, stream>>>(cw, weff);
  k_dscale  <<<dim3(32),    dim3(256), 0, stream>>>(s, wsum, dscale);
  k_xs      <<<dim3(36992), dim3(256), 0, stream>>>(x, s, xs);
  k_conv_wmma<<<dim3(NGEMM / 128, 4, 4), dim3(256), 0, stream>>>(
      weff, xs, dscale, noise, noise_w, act_b, out);
}
